// SwinDecoderBlock_609885356543
// MI455X (gfx1250) — compile-verified, matrix operand order checked by reading, not run on hardware
//
#include <hip/hip_runtime.h>
#include <hip/hip_bf16.h>

// ---------------------------------------------------------------------------
// Swin decoder block for MI455X (gfx1250), wave32 + WMMA f32_16x16x32_f16.
// ---------------------------------------------------------------------------

typedef __attribute__((ext_vector_type(16))) _Float16 v16h;
typedef __attribute__((ext_vector_type(8)))  float    v8f;
typedef _Float16 h16;

#define WMMA16(a, b, c) \
  __builtin_amdgcn_wmma_f32_16x16x32_f16(false, (a), false, (b), (short)0, (c), false, false)

// A-fragment (and Bt-fragment) loader: matrix stored row-major with the
// "fragment row" contiguous in K.  Lane l holds row (l&15); K layout per the
// CDNA5 ISA 16-bit A-matrix table: VGPRs 0-3 hold K = half*8 + 0..7,
// VGPRs 4-7 hold K = 16 + half*8 + 0..7  (half = l>>4).
static __device__ __forceinline__ v16h load_frag_h(const h16* base, int ld) {
  const int lane = threadIdx.x & 31;
  const h16* p = base + (size_t)(lane & 15) * ld + ((lane >> 4) << 3);
  v16h f;
  ((float4*)&f)[0] = *(const float4*)(p);
  ((float4*)&f)[1] = *(const float4*)(p + 16);
  return f;
}

static __device__ __forceinline__ v16h load_frag_f32(const float* base, int ld) {
  const int lane = threadIdx.x & 31;
  const float* p = base + (size_t)(lane & 15) * ld + ((lane >> 4) << 3);
  v16h f;
#pragma unroll
  for (int i = 0; i < 8; ++i) { f[i] = (h16)p[i]; f[i + 8] = (h16)p[i + 16]; }
  return f;
}

// Shared GEMM mainloop: workgroup = 256 threads (8 waves), tile 64(M) x 256(N).
// Wave w owns columns [n0 + 32w, n0 + 32w + 32).  A tile (64x32 f16) staged in
// LDS each K-step; Bt is [N][K] row-major in global (L2-hot weights).
template <typename RowMap>
static __device__ __forceinline__ void gemm_mainloop(
    const h16* __restrict__ A, int lda, const h16* __restrict__ Bt, int K,
    h16* As, int n0, v8f (&acc)[4][2], RowMap rowmap) {
  const v8f vz = {};
#pragma unroll
  for (int i = 0; i < 4; ++i)
#pragma unroll
    for (int j = 0; j < 2; ++j) acc[i][j] = vz;

  const int tid  = threadIdx.x;
  const int wave = tid >> 5;
  const int col0 = n0 + wave * 32;
  const int lrow = tid >> 2;          // 0..63  (staged row)
  const int lcol = (tid & 3) << 3;    // 0,8,16,24
  const size_t arow = (size_t)rowmap(lrow) * (size_t)lda;

  for (int k0 = 0; k0 < K; k0 += 32) {
    __syncthreads();
    *(float4*)&As[lrow * 32 + lcol] = *(const float4*)&A[arow + k0 + lcol];
    __syncthreads();
    v16h b0 = load_frag_h(Bt + (size_t)col0 * K + k0, K);
    v16h b1 = load_frag_h(Bt + (size_t)(col0 + 16) * K + k0, K);
#pragma unroll
    for (int i = 0; i < 4; ++i) {
      v16h a = load_frag_h(As + i * 16 * 32, 32);
      acc[i][0] = WMMA16(a, b0, acc[i][0]);
      acc[i][1] = WMMA16(a, b1, acc[i][1]);
    }
  }
}

// ------------------------------- prep kernels ------------------------------

__global__ void k_cvt(const float* __restrict__ src, h16* __restrict__ dst, int n) {
  int i = blockIdx.x * 256 + threadIdx.x;
  if (i < n) dst[i] = (h16)src[i];
}

// up_w [cin=512][co=256][kh=2][kw=2] -> Bt[n = (kh*2+kw)*256+co][k = cin]
__global__ void k_upw(const float* __restrict__ upw, h16* __restrict__ dst) {
  int i = blockIdx.x * 256 + threadIdx.x;
  if (i < 512 * 1024) {
    int cin = i >> 10, n = i & 1023;
    int kh = n >> 9, kw = (n >> 8) & 1, co = n & 255;
    dst[(size_t)n * 512 + cin] = (h16)upw[((cin * 256 + co) * 2 + kh) * 2 + kw];
  }
}

// rel_bias_table [225][8] -> biasx[h][m][n]
__global__ void k_bias(const float* __restrict__ table, float* __restrict__ biasx) {
  int i = blockIdx.x * 256 + threadIdx.x;
  if (i < 8 * 64 * 64) {
    int h = i >> 12, m = (i >> 6) & 63, n = i & 63;
    int my = m >> 3, mx = m & 7, ny = n >> 3, nx = n & 7;
    int idx = (my - ny + 7) * 15 + (mx - nx + 7);
    biasx[i] = table[idx * 8 + h];
  }
}

// x [8,512,64,64] f32 -> xh [8*4096][512] f16 (token-major)
__global__ void k_xpose_x(const float* __restrict__ x, h16* __restrict__ xh) {
  size_t i = (size_t)blockIdx.x * 256 + threadIdx.x;
  if (i < (size_t)8 * 512 * 4096) {
    size_t P = i >> 9; int c = (int)(i & 511);
    int b = (int)(P >> 12); int p = (int)(P & 4095);
    xh[i] = (h16)x[(((size_t)b * 512 + c) << 12) + p];
  }
}

// skip_feat [8,256,128,128] f32 -> skiph [131072][256] f16
__global__ void k_xpose_skip(const float* __restrict__ s, h16* __restrict__ sh) {
  size_t i = (size_t)blockIdx.x * 256 + threadIdx.x;
  if (i < (size_t)8 * 256 * 16384) {
    size_t T = i >> 8; int c = (int)(i & 255);
    int b = (int)(T >> 14); int pos = (int)(T & 16383);
    sh[i] = (h16)s[(((size_t)b * 256 + c) << 14) + pos];
  }
}

// ------------------------- upconv (ConvTranspose2d) ------------------------
// out[pixel P][n=(kh,kw,co)] = x_h[P] . up_wh[n]; scatter to shortcut tokens.
__global__ void __launch_bounds__(256)
k_upconv(const h16* __restrict__ xh, const h16* __restrict__ upwh,
         const float* __restrict__ upb, float* __restrict__ shortcut) {
  extern __shared__ __align__(16) char smem[];
  h16* As = (h16*)smem;
  const int mb = blockIdx.x;              // 512 blocks of 64 input pixels
  const int n0 = blockIdx.y * 256;        // 0..1023
  v8f acc[4][2];
  gemm_mainloop(xh + (size_t)mb * 64 * 512, 512, upwh, 512, As, n0, acc,
                [](int r) { return r; });
  const int lane = threadIdx.x & 31, wave = threadIdx.x >> 5;
  const int half = lane >> 4, nn = lane & 15;
#pragma unroll
  for (int i = 0; i < 4; ++i)
#pragma unroll
    for (int j = 0; j < 2; ++j)
#pragma unroll
      for (int r = 0; r < 8; ++r) {
        int m = i * 16 + r + 8 * half;
        int n = n0 + wave * 32 + j * 16 + nn;
        int P = mb * 64 + m;
        int b = P >> 12, p = P & 4095;
        int hh = p >> 6, ww = p & 63;
        int kh = n >> 9, kw = (n >> 8) & 1, co = n & 255;
        size_t tok = ((size_t)b << 14) + (size_t)(hh * 2 + kh) * 128 + (ww * 2 + kw);
        shortcut[tok * 256 + co] = acc[i][j][r] + upb[co];
      }
}

// ------------------- skip adapter + residual + LayerNorm1 ------------------
__global__ void __launch_bounds__(256)
k_skipln(const h16* __restrict__ skiph, const h16* __restrict__ skw,
         const float* __restrict__ shortcut, const float* __restrict__ g,
         const float* __restrict__ bta, h16* __restrict__ xnorm) {
  extern __shared__ __align__(16) char smem[];
  h16*   As   = (h16*)smem;                       // 4 KB
  float* Ysh  = (float*)(smem + 4096);            // 64 KB
  float* red  = (float*)(smem + 4096 + 65536);    // 2 KB
  float* musd = red + 512;                        // 0.5 KB
  const int mb = blockIdx.x;                      // 2048 blocks of 64 tokens
  v8f acc[4][2];
  gemm_mainloop(skiph + (size_t)mb * 64 * 256, 256, skw, 256, As, 0, acc,
                [](int r) { return r; });
  const int lane = threadIdx.x & 31, wave = threadIdx.x >> 5;
  const int half = lane >> 4, nn = lane & 15;
#pragma unroll
  for (int i = 0; i < 4; ++i)
#pragma unroll
    for (int j = 0; j < 2; ++j)
#pragma unroll
      for (int r = 0; r < 8; ++r) {
        int m = i * 16 + r + 8 * half;
        int n = wave * 32 + j * 16 + nn;
        size_t tok = (size_t)mb * 64 + m;
        Ysh[m * 256 + n] = acc[i][j][r] + shortcut[tok * 256 + n];
      }
  __syncthreads();
  const int t = threadIdx.x;
  {
    int row = t & 63, seg = t >> 6;
    float s = 0.f, q = 0.f;
    for (int c = seg * 64; c < seg * 64 + 64; ++c) {
      float v = Ysh[row * 256 + c]; s += v; q += v * v;
    }
    red[seg * 64 + row] = s;
    red[256 + seg * 64 + row] = q;
  }
  __syncthreads();
  if (t < 64) {
    float s = red[t] + red[64 + t] + red[128 + t] + red[192 + t];
    float q = red[256 + t] + red[320 + t] + red[384 + t] + red[448 + t];
    float mu = s * (1.f / 256.f);
    float var = q * (1.f / 256.f) - mu * mu;
    musd[t * 2] = mu;
    musd[t * 2 + 1] = rsqrtf(var + 1e-6f);
  }
  __syncthreads();
  {
    int row = t & 63, seg = t >> 6;
    float mu = musd[row * 2], rs = musd[row * 2 + 1];
    size_t tok = (size_t)mb * 64 + row;
    for (int c = seg * 64; c < seg * 64 + 64; ++c) {
      float v = (Ysh[row * 256 + c] - mu) * rs * g[c] + bta[c];
      xnorm[tok * 256 + c] = (h16)v;
    }
  }
}

// ------------------------------- QKV GEMM ----------------------------------
// M-block = one 8x8 window (64 tokens gathered from token-major xnorm).
__global__ void __launch_bounds__(256)
k_qkv(const h16* __restrict__ xnorm, const h16* __restrict__ qkvw,
      const float* __restrict__ qkvb, h16* __restrict__ qkvbuf) {
  extern __shared__ __align__(16) char smem[];
  h16* As = (h16*)smem;
  const int wd = blockIdx.x;              // 2048 windows
  const int n0 = blockIdx.y * 256;        // 0..767
  const int b = wd >> 8, rem = wd & 255, nh = rem >> 4, nw = rem & 15;
  auto rowmap = [b, nh, nw](int m) {
    int r = m >> 3, c = m & 7;
    return (b << 14) + (nh * 8 + r) * 128 + nw * 8 + c;
  };
  v8f acc[4][2];
  gemm_mainloop(xnorm, 256, qkvw, 256, As, n0, acc, rowmap);
  const int lane = threadIdx.x & 31, wave = threadIdx.x >> 5;
  const int half = lane >> 4, nn = lane & 15;
#pragma unroll
  for (int i = 0; i < 4; ++i)
#pragma unroll
    for (int j = 0; j < 2; ++j)
#pragma unroll
      for (int r = 0; r < 8; ++r) {
        int m = i * 16 + r + 8 * half;
        int n = n0 + wave * 32 + j * 16 + nn;
        int which = n >> 8, head = (n >> 5) & 7, d = n & 31;
        float v = acc[i][j][r] + qkvb[n];
        if (which == 0) v *= 0.17677669529663687f;   // 1/sqrt(32)
        size_t base = (((size_t)wd * 3 + which) * 8 + head) * 2048;
        if (which == 2) qkvbuf[base + (size_t)d * 64 + m] = (h16)v;  // V^T
        else            qkvbuf[base + (size_t)m * 32 + d] = (h16)v;
      }
}

// ---------------------- windowed attention (per head) ----------------------
__global__ void __launch_bounds__(256)
k_attn(const h16* __restrict__ qkvbuf, const float* __restrict__ biasx,
       h16* __restrict__ obuf) {
  extern __shared__ __align__(16) char smem[];   // 8 heads * 64*64 f32 = 128KB
  const int wd = blockIdx.x;
  const int wave = threadIdx.x >> 5;             // head
  const int lane = threadIdx.x & 31;
  const int half = lane >> 4, nn = lane & 15;
  float* S = (float*)smem + (size_t)wave * 64 * 64;
  const h16* q  = qkvbuf + (((size_t)wd * 3 + 0) * 8 + wave) * 2048;
  const h16* kk = qkvbuf + (((size_t)wd * 3 + 1) * 8 + wave) * 2048;
  const h16* vv = qkvbuf + (((size_t)wd * 3 + 2) * 8 + wave) * 2048;
  const float* bias = biasx + (size_t)wave * 64 * 64;

  // S = (q*scale) k^T  + bias
  v16h af[4], bf[4];
#pragma unroll
  for (int i = 0; i < 4; ++i) af[i] = load_frag_h(q + i * 16 * 32, 32);
#pragma unroll
  for (int j = 0; j < 4; ++j) bf[j] = load_frag_h(kk + j * 16 * 32, 32);
#pragma unroll
  for (int i = 0; i < 4; ++i)
#pragma unroll
    for (int j = 0; j < 4; ++j) {
      v8f a = {};
      a = WMMA16(af[i], bf[j], a);
#pragma unroll
      for (int r = 0; r < 8; ++r) {
        int m = i * 16 + r + 8 * half;
        int n = j * 16 + nn;
        S[m * 64 + n] = a[r] + bias[m * 64 + n];
      }
    }
  __syncthreads();
  // softmax over keys: 2 rows per lane
#pragma unroll
  for (int rr = 0; rr < 2; ++rr) {
    int row = lane * 2 + rr;
    float mx = -1e30f;
    for (int c = 0; c < 64; ++c) mx = fmaxf(mx, S[row * 64 + c]);
    float sum = 0.f;
    for (int c = 0; c < 64; ++c) {
      float e = __expf(S[row * 64 + c] - mx);
      S[row * 64 + c] = e; sum += e;
    }
    float inv = 1.f / sum;
    for (int c = 0; c < 64; ++c) S[row * 64 + c] *= inv;
  }
  __syncthreads();
  // O = P V   (P f32 in LDS -> f16 frags; V stored transposed [32][64])
  v8f oacc[4][2];
  const v8f vz = {};
#pragma unroll
  for (int i = 0; i < 4; ++i) { oacc[i][0] = vz; oacc[i][1] = vz; }
  for (int kq = 0; kq < 64; kq += 32) {
    v16h vb0 = load_frag_h(vv + 0 * 16 * 64 + kq, 64);
    v16h vb1 = load_frag_h(vv + 1 * 16 * 64 + kq, 64);
#pragma unroll
    for (int i = 0; i < 4; ++i) {
      v16h pa = load_frag_f32(S + (i * 16) * 64 + kq, 64);
      oacc[i][0] = WMMA16(pa, vb0, oacc[i][0]);
      oacc[i][1] = WMMA16(pa, vb1, oacc[i][1]);
    }
  }
#pragma unroll
  for (int i = 0; i < 4; ++i)
#pragma unroll
    for (int j = 0; j < 2; ++j)
#pragma unroll
      for (int r = 0; r < 8; ++r) {
        int m = i * 16 + r + 8 * half;
        int d = j * 16 + nn;
        obuf[((size_t)wd * 64 + m) * 256 + wave * 32 + d] = (h16)oacc[i][j][r];
      }
}

// ----------------- proj + shortcut residual + LayerNorm2 -------------------
__global__ void __launch_bounds__(256)
k_projln(const h16* __restrict__ obuf, const h16* __restrict__ projw,
         const float* __restrict__ projb, const float* __restrict__ shortcut,
         const float* __restrict__ g2, const float* __restrict__ b2n,
         float* __restrict__ xr, h16* __restrict__ xn2) {
  extern __shared__ __align__(16) char smem[];
  h16*   As   = (h16*)smem;
  float* Ysh  = (float*)(smem + 4096);
  float* red  = (float*)(smem + 4096 + 65536);
  float* musd = red + 512;
  const int wd = blockIdx.x;
  const int b = wd >> 8, rem = wd & 255, nh = rem >> 4, nw = rem & 15;
  v8f acc[4][2];
  gemm_mainloop(obuf + (size_t)wd * 64 * 256, 256, projw, 256, As, 0, acc,
                [](int r) { return r; });
  const int lane = threadIdx.x & 31, wave = threadIdx.x >> 5;
  const int half = lane >> 4, nn = lane & 15;
#pragma unroll
  for (int i = 0; i < 4; ++i)
#pragma unroll
    for (int j = 0; j < 2; ++j)
#pragma unroll
      for (int r = 0; r < 8; ++r) {
        int m = i * 16 + r + 8 * half;
        int n = wave * 32 + j * 16 + nn;
        int r8 = m >> 3, c8 = m & 7;
        size_t tok = ((size_t)b << 14) + (size_t)(nh * 8 + r8) * 128 + nw * 8 + c8;
        float v = acc[i][j][r] + projb[n] + shortcut[tok * 256 + n];
        Ysh[m * 256 + n] = v;
        xr[tok * 256 + n] = v;
      }
  __syncthreads();
  const int t = threadIdx.x;
  {
    int row = t & 63, seg = t >> 6;
    float s = 0.f, q = 0.f;
    for (int c = seg * 64; c < seg * 64 + 64; ++c) {
      float v = Ysh[row * 256 + c]; s += v; q += v * v;
    }
    red[seg * 64 + row] = s;
    red[256 + seg * 64 + row] = q;
  }
  __syncthreads();
  if (t < 64) {
    float s = red[t] + red[64 + t] + red[128 + t] + red[192 + t];
    float q = red[256 + t] + red[320 + t] + red[384 + t] + red[448 + t];
    float mu = s * (1.f / 256.f);
    float var = q * (1.f / 256.f) - mu * mu;
    musd[t * 2] = mu;
    musd[t * 2 + 1] = rsqrtf(var + 1e-6f);
  }
  __syncthreads();
  {
    int row = t & 63, seg = t >> 6;
    float mu = musd[row * 2], rs = musd[row * 2 + 1];
    int r8 = row >> 3, c8 = row & 7;
    size_t tok = ((size_t)b << 14) + (size_t)(nh * 8 + r8) * 128 + nw * 8 + c8;
    for (int c = seg * 64; c < seg * 64 + 64; ++c) {
      float v = (Ysh[row * 256 + c] - mu) * rs * g2[c] + b2n[c];
      xn2[tok * 256 + c] = (h16)v;
    }
  }
}

// ------------------------------- MLP layers --------------------------------
__global__ void __launch_bounds__(256)
k_mlp1(const h16* __restrict__ xn2, const h16* __restrict__ w1,
       const float* __restrict__ b1, h16* __restrict__ hbuf) {
  extern __shared__ __align__(16) char smem[];
  h16* As = (h16*)smem;
  const int mb = blockIdx.x;              // 2048 token blocks
  const int n0 = blockIdx.y * 256;        // 0..1023
  v8f acc[4][2];
  gemm_mainloop(xn2 + (size_t)mb * 64 * 256, 256, w1, 256, As, n0, acc,
                [](int r) { return r; });
  const int lane = threadIdx.x & 31, wave = threadIdx.x >> 5;
  const int half = lane >> 4, nn = lane & 15;
#pragma unroll
  for (int i = 0; i < 4; ++i)
#pragma unroll
    for (int j = 0; j < 2; ++j)
#pragma unroll
      for (int r = 0; r < 8; ++r) {
        int m = i * 16 + r + 8 * half;
        int n = n0 + wave * 32 + j * 16 + nn;
        float v = acc[i][j][r] + b1[n];
        v = 0.5f * v * (1.f + erff(v * 0.70710678118654752f));   // exact GELU
        hbuf[((size_t)mb * 64 + m) * 1024 + n] = (h16)v;
      }
}

__global__ void __launch_bounds__(256)
k_mlp2(const h16* __restrict__ hbuf, const h16* __restrict__ w2,
       const float* __restrict__ b2, const float* __restrict__ xr,
       float* __restrict__ out) {
  extern __shared__ __align__(16) char smem[];
  h16* As = (h16*)smem;
  const int mb = blockIdx.x;
  v8f acc[4][2];
  gemm_mainloop(hbuf + (size_t)mb * 64 * 1024, 1024, w2, 1024, As, 0, acc,
                [](int r) { return r; });
  const int lane = threadIdx.x & 31, wave = threadIdx.x >> 5;
  const int half = lane >> 4, nn = lane & 15;
#pragma unroll
  for (int i = 0; i < 4; ++i)
#pragma unroll
    for (int j = 0; j < 2; ++j)
#pragma unroll
      for (int r = 0; r < 8; ++r) {
        int m = i * 16 + r + 8 * half;
        int n = wave * 32 + j * 16 + nn;
        size_t tok = (size_t)mb * 64 + m;
        float v = acc[i][j][r] + b2[n] + xr[tok * 256 + n];
        int b = (int)(tok >> 14), pos = (int)(tok & 16383);
        out[(((size_t)b * 256 + n) << 14) + pos] = v;   // NCHW
      }
}

// ------------------------------- host launch -------------------------------

extern "C" void kernel_launch(void* const* d_in, const int* in_sizes, int n_in,
                              void* d_out, int out_size, void* d_ws, size_t ws_size,
                              hipStream_t stream) {
  const float* x      = (const float*)d_in[0];
  const float* skipf  = (const float*)d_in[1];
  const float* up_w   = (const float*)d_in[2];
  const float* up_b   = (const float*)d_in[3];
  const float* skip_w = (const float*)d_in[4];
  const float* n1g    = (const float*)d_in[5];
  const float* n1b    = (const float*)d_in[6];
  const float* n2g    = (const float*)d_in[7];
  const float* n2b    = (const float*)d_in[8];
  const float* qkv_w  = (const float*)d_in[9];
  const float* qkv_b  = (const float*)d_in[10];
  const float* proj_w = (const float*)d_in[11];
  const float* proj_b = (const float*)d_in[12];
  const float* rbt    = (const float*)d_in[13];
  const float* mlp_w1 = (const float*)d_in[14];
  const float* mlp_b1 = (const float*)d_in[15];
  const float* mlp_w2 = (const float*)d_in[16];
  const float* mlp_b2 = (const float*)d_in[17];
  float* out = (float*)d_out;

  char* W = (char*)d_ws;
  size_t off = 0;
  auto take = [&](size_t bytes) {
    size_t r = off; off += (bytes + 255) & ~(size_t)255; return r;
  };
  h16*   upwh   = (h16*)(W + take((size_t)512 * 1024 * 2));
  h16*   skwh   = (h16*)(W + take((size_t)256 * 256 * 2));
  h16*   qkvwh  = (h16*)(W + take((size_t)768 * 256 * 2));
  h16*   prjwh  = (h16*)(W + take((size_t)256 * 256 * 2));
  h16*   w1h    = (h16*)(W + take((size_t)1024 * 256 * 2));
  h16*   w2h    = (h16*)(W + take((size_t)256 * 1024 * 2));
  float* biasx  = (float*)(W + take((size_t)8 * 64 * 64 * 4));
  h16*   xh     = (h16*)(W + take((size_t)32768 * 512 * 2));
  h16*   skiph  = (h16*)(W + take((size_t)131072 * 256 * 2));   // reused as xn2
  h16*   xn2    = skiph;                                        // (lifetimes disjoint)
  float* shortcut = (float*)(W + take((size_t)131072 * 256 * 4));
  h16*   xnorm  = (h16*)(W + take((size_t)131072 * 256 * 2));
  char*  QH     = W + take((size_t)131072 * 1024 * 2);          // qkvbuf then hbuf
  h16*   qkvbuf = (h16*)QH;
  h16*   hbuf   = (h16*)QH;
  h16*   obuf   = (h16*)(W + take((size_t)2048 * 64 * 256 * 2));
  float* xr     = (float*)(W + take((size_t)131072 * 256 * 4));
  (void)ws_size; (void)n_in; (void)in_sizes; (void)out_size;

  // --- prep: weight converts / reorders, bias expand, activation transposes
  k_upw<<<2048, 256, 0, stream>>>(up_w, upwh);
  k_cvt<<<(65536 + 255) / 256, 256, 0, stream>>>(skip_w, skwh, 65536);
  k_cvt<<<(196608 + 255) / 256, 256, 0, stream>>>(qkv_w, qkvwh, 196608);
  k_cvt<<<(65536 + 255) / 256, 256, 0, stream>>>(proj_w, prjwh, 65536);
  k_cvt<<<(262144 + 255) / 256, 256, 0, stream>>>(mlp_w1, w1h, 262144);
  k_cvt<<<(262144 + 255) / 256, 256, 0, stream>>>(mlp_w2, w2h, 262144);
  k_bias<<<128, 256, 0, stream>>>(rbt, biasx);
  k_xpose_x<<<65536, 256, 0, stream>>>(x, xh);
  k_xpose_skip<<<131072, 256, 0, stream>>>(skipf, skiph);

  const size_t SM_GEMM = 4096;
  const size_t SM_LN   = 4096 + 65536 + 2560;
  const size_t SM_ATTN = (size_t)8 * 64 * 64 * 4;

  // --- up-conv (ConvTranspose2d as GEMM) -> shortcut
  k_upconv<<<dim3(512, 4), 256, SM_GEMM, stream>>>(xh, upwh, up_b, shortcut);
  // --- skip adapter + residual + LN1 -> xnorm (f16)
  k_skipln<<<2048, 256, SM_LN, stream>>>(skiph, skwh, shortcut, n1g, n1b, xnorm);
  // --- qkv projection into per-(window,head) layout
  k_qkv<<<dim3(2048, 3), 256, SM_GEMM, stream>>>(xnorm, qkvwh, qkv_b, qkvbuf);
  // --- windowed attention
  k_attn<<<2048, 256, SM_ATTN, stream>>>(qkvbuf, biasx, obuf);
  // --- proj + residual + LN2 -> xr (f32), xn2 (f16)
  k_projln<<<2048, 256, SM_LN, stream>>>(obuf, prjwh, proj_b, shortcut,
                                         n2g, n2b, xr, xn2);
  // --- MLP
  k_mlp1<<<dim3(2048, 4), 256, SM_GEMM, stream>>>(xn2, w1h, mlp_b1, hbuf);
  k_mlp2<<<2048, 256, SM_GEMM, stream>>>(hbuf, w2h, mlp_b2, xr, out);
}
